// DattaBotModel_78323023610501
// MI455X (gfx1250) — compile-verified, matrix operand order checked by reading, not run on hardware
//
#include <hip/hip_runtime.h>
#include <math.h>

// ---------------- model dims ----------------
#define BATCH 2
#define SEQ   1024
#define DMODEL 1024
#define NHEADS 16
#define HEADD  64
#define NEXP   8
#define TOPK   2
#define HHID   4096
#define NTOK   (BATCH * SEQ)          // 2048
#define PAIR_CAP 5120                 // 2*NTOK + 8*127, rounded to 128-multiple

typedef __attribute__((ext_vector_type(16))) __bf16 v16bf;
typedef __attribute__((ext_vector_type(8)))  float  v8f;

// flags for generic GEMM
#define FLAG_GELU    1
#define FLAG_OUTBF   2
#define FLAG_SCATTER 4
#define FLAG_RES     8

// ---------------- helpers ----------------
__device__ __forceinline__ unsigned short f2bf(float f) {
    union { float f; unsigned u; } c; c.f = f;
    unsigned u = c.u;
    unsigned r = (u + 0x7FFFu + ((u >> 16) & 1u)) >> 16;   // RNE
    return (unsigned short)r;
}
__device__ __forceinline__ float bf2f(unsigned short h) {
    union { unsigned u; float f; } c; c.u = ((unsigned)h) << 16;
    return c.f;
}

__device__ __forceinline__ v8f wmma_bf16(v16bf a, v16bf b, v8f c) {
    // v_wmma_f32_16x16x32_bf16 : D = A(16x32) * B(32x16) + C(16x16 f32)
    return __builtin_amdgcn_wmma_f32_16x16x32_bf16(
        false, a, false, b, (short)0, c, false, false);
}

// Async copy 16B global -> LDS (ASYNCcnt-tracked, CDNA5 ISA).
// Flat LDS addresses carry the LDS byte offset in addr[31:0] (aperture decode),
// so truncating the generic pointer yields the VDST LDS-address operand.
__device__ __forceinline__ void async_copy_b128(unsigned lds_off, const void* gsrc) {
    unsigned long long ga = (unsigned long long)(size_t)gsrc;
    asm volatile("global_load_async_to_lds_b128 %0, %1, off"
                 :: "v"(lds_off), "v"(ga) : "memory");
}
__device__ __forceinline__ void wait_async0() {
    asm volatile("s_wait_asynccnt 0x0" ::: "memory");
}
__device__ __forceinline__ unsigned lds_addr(const void* p) {
    return (unsigned)(size_t)p;
}

// Load a 16-element bf16 fragment for lane (hl = lane>>4) from a 32-wide
// K-chunk starting at rowPtr.  Per ISA 16-bit A layout: lane half selects
// K = hl*8..hl*8+7  and  K = 16+hl*8..16+hl*8+7.
__device__ __forceinline__ v16bf load_frag(const unsigned short* rowPtr, int hl) {
    union { v16bf v; uint4 q[2]; } u;
    u.q[0] = *reinterpret_cast<const uint4*>(rowPtr + hl * 8);
    u.q[1] = *reinterpret_cast<const uint4*>(rowPtr + 16 + hl * 8);
    return u.v;
}

// ---------------- RMSNorm (f32 in -> bf16 out) ----------------
__global__ __launch_bounds__(256)
void rmsnorm_kernel(const float* __restrict__ x, const float* __restrict__ w,
                    unsigned short* __restrict__ out, int D) {
    int t = blockIdx.x;
    const float* row = x + (size_t)t * D;
    float s = 0.f;
    for (int i = threadIdx.x; i < D; i += blockDim.x) { float v = row[i]; s += v * v; }
    __shared__ float red[256];
    red[threadIdx.x] = s;
    __syncthreads();
    for (int off = 128; off > 0; off >>= 1) {
        if ((int)threadIdx.x < off) red[threadIdx.x] += red[threadIdx.x + off];
        __syncthreads();
    }
    float inv = rsqrtf(red[0] / (float)D + 1e-6f);
    for (int i = threadIdx.x; i < D; i += blockDim.x)
        out[(size_t)t * D + i] = f2bf(row[i] * inv * w[i]);
}

// ---------------- RoPE on q,k (bf16 [T][D], col = h*64+d) ----------------
__global__ __launch_bounds__(256)
void rope_kernel(unsigned short* __restrict__ q, unsigned short* __restrict__ k) {
    int idx = blockIdx.x * blockDim.x + threadIdx.x;      // T*NH*32
    if (idx >= NTOK * NHEADS * 32) return;
    int d = idx & 31;
    int h = (idx >> 5) & (NHEADS - 1);
    int t = idx >> 9;
    int s = t & (SEQ - 1);
    float inv_freq = __powf(10000.0f, -(float)(2 * d) / 64.0f);
    float ang = (float)s * inv_freq;
    float c = __cosf(ang), sn = __sinf(ang);
    size_t base = (size_t)t * DMODEL + h * HEADD + d;
    float q1 = bf2f(q[base]), q2 = bf2f(q[base + 32]);
    q[base]      = f2bf(q1 * c - q2 * sn);
    q[base + 32] = f2bf(q2 * c + q1 * sn);
    float k1 = bf2f(k[base]), k2 = bf2f(k[base + 32]);
    k[base]      = f2bf(k1 * c - k2 * sn);
    k[base + 32] = f2bf(k2 * c + k1 * sn);
}

// ---------------- 128x64-tile bf16 WMMA GEMM, double-buffered ----------------
// out[M,N] = act( A_bf16[M,Kd] @ W_f32[N,Kd]^T + bias )  (+residual / scatter)
// 256 threads = 8 waves; wave w owns rows w*16..+15 x all 64 cols (4 accums).
// K-step 64.  A staged via GLOBAL_LOAD_ASYNC_TO_LDS_B128 into the next buffer
// while WMMAs consume the current; W staged through registers (f32->bf16).
// All 10 fragments of a K-step are preloaded into distinct registers so the
// ds_load_b128s overlap the WMMA issue instead of serializing on dscnt==0.
__global__ __launch_bounds__(256)
void gemm_bf16_wmma(const unsigned short* __restrict__ A, int lda,
                    const float* __restrict__ W, int ldw,
                    const float* __restrict__ bias,
                    const float* __restrict__ residual,
                    float* __restrict__ outF, unsigned short* __restrict__ outB,
                    int M, int N, int Kd,
                    const int* __restrict__ rowmap,        // gather A rows (-1 => zero)
                    const int* __restrict__ tile_expert,   // per-row expert id (or null)
                    size_t w_estride, size_t b_estride,
                    const int* __restrict__ scat_tok, const float* __restrict__ scat_w,
                    int flags) {
    int nBase = blockIdx.x * 64;
    int mBase = blockIdx.y * 128;
    if (tile_expert) {
        int e = tile_expert[mBase];                // 128-row tile is expert-uniform
        W += (size_t)e * w_estride;
        if (bias) bias += (size_t)e * b_estride;
    }
    __shared__ alignas(32) unsigned short As[2][128][64];   // 32 KB
    __shared__ alignas(32) unsigned short Ws[2][64][64];    // 16 KB

    int tid = threadIdx.x, wave = tid >> 5, lane = tid & 31;
    int hl = lane >> 4, ln = lane & 15;

    v8f acc[4];
    const v8f z8 = {0.f,0.f,0.f,0.f,0.f,0.f,0.f,0.f};
    for (int i = 0; i < 4; i++) acc[i] = z8;

    // A staging: thread -> row tid>>1 (128 rows), cols (tid&1)*32..+31 (4x b128)
    int arow = tid >> 1;
    int ac0  = (tid & 1) * 32;
    long asrc;
    if (rowmap) { int g = rowmap[mBase + arow]; asrc = (g < 0) ? -1 : (long)g * lda; }
    else        { asrc = (long)(mBase + arow) * lda; }
    // W staging: thread -> row tid>>2 (64 rows), k (tid&3)*16..+15
    int wrow = tid >> 2;
    int wk0  = (tid & 3) * 16;
    const float* wRowPtr = W + (size_t)(nBase + wrow) * ldw + wk0;

    const int nk = Kd >> 6;

    // ---- stage chunk 0 into buffer 0 ----
    {
        if (asrc < 0) {
            uint4 z = {0u,0u,0u,0u};
            uint4* d = reinterpret_cast<uint4*>(&As[0][arow][ac0]);
            d[0] = z; d[1] = z; d[2] = z; d[3] = z;
        } else {
            const unsigned short* src = A + asrc + ac0;
            for (int i = 0; i < 4; i++)
                async_copy_b128(lds_addr(&As[0][arow][ac0 + i * 8]), src + i * 8);
        }
        float4 f0 = reinterpret_cast<const float4*>(wRowPtr)[0];
        float4 f1 = reinterpret_cast<const float4*>(wRowPtr)[1];
        float4 f2 = reinterpret_cast<const float4*>(wRowPtr)[2];
        float4 f3 = reinterpret_cast<const float4*>(wRowPtr)[3];
        uint4 p0, p1;
        p0.x = (unsigned)f2bf(f0.x) | ((unsigned)f2bf(f0.y) << 16);
        p0.y = (unsigned)f2bf(f0.z) | ((unsigned)f2bf(f0.w) << 16);
        p0.z = (unsigned)f2bf(f1.x) | ((unsigned)f2bf(f1.y) << 16);
        p0.w = (unsigned)f2bf(f1.z) | ((unsigned)f2bf(f1.w) << 16);
        p1.x = (unsigned)f2bf(f2.x) | ((unsigned)f2bf(f2.y) << 16);
        p1.y = (unsigned)f2bf(f2.z) | ((unsigned)f2bf(f2.w) << 16);
        p1.z = (unsigned)f2bf(f3.x) | ((unsigned)f2bf(f3.y) << 16);
        p1.w = (unsigned)f2bf(f3.z) | ((unsigned)f2bf(f3.w) << 16);
        uint4* d = reinterpret_cast<uint4*>(&Ws[0][wrow][wk0]);
        d[0] = p0; d[1] = p1;
        wait_async0();
        __syncthreads();
    }

    for (int i = 0; i < nk; i++) {
        int cur = i & 1, nxt = cur ^ 1;
        bool haveNext = (i + 1 < nk);
        int k1 = (i + 1) << 6;

        // issue next A chunk (async -> LDS[nxt]) and next W loads (regs)
        float4 f0, f1, f2, f3;
        if (haveNext) {
            if (asrc < 0) {
                uint4 z = {0u,0u,0u,0u};
                uint4* d = reinterpret_cast<uint4*>(&As[nxt][arow][ac0]);
                d[0] = z; d[1] = z; d[2] = z; d[3] = z;
            } else {
                const unsigned short* src = A + asrc + k1 + ac0;
                for (int j = 0; j < 4; j++)
                    async_copy_b128(lds_addr(&As[nxt][arow][ac0 + j * 8]), src + j * 8);
            }
            const float* wp = wRowPtr + k1;
            f0 = reinterpret_cast<const float4*>(wp)[0];
            f1 = reinterpret_cast<const float4*>(wp)[1];
            f2 = reinterpret_cast<const float4*>(wp)[2];
            f3 = reinterpret_cast<const float4*>(wp)[3];
        }

        // ---- compute: preload all fragments, then 8 WMMAs per wave ----
        {
            const unsigned short* aRow = &As[cur][wave * 16 + ln][0];
            v16bf af[2];
            af[0] = load_frag(aRow, hl);
            af[1] = load_frag(aRow + 32, hl);
            v16bf bf[2][4];
#pragma unroll
            for (int sub = 0; sub < 2; sub++)
#pragma unroll
                for (int nt = 0; nt < 4; nt++)
                    bf[sub][nt] = load_frag(&Ws[cur][nt * 16 + ln][sub * 32], hl);
#pragma unroll
            for (int sub = 0; sub < 2; sub++)
#pragma unroll
                for (int nt = 0; nt < 4; nt++)
                    acc[nt] = wmma_bf16(af[sub], bf[sub][nt], acc[nt]);
        }

        if (haveNext) {
            uint4 p0, p1;
            p0.x = (unsigned)f2bf(f0.x) | ((unsigned)f2bf(f0.y) << 16);
            p0.y = (unsigned)f2bf(f0.z) | ((unsigned)f2bf(f0.w) << 16);
            p0.z = (unsigned)f2bf(f1.x) | ((unsigned)f2bf(f1.y) << 16);
            p0.w = (unsigned)f2bf(f1.z) | ((unsigned)f2bf(f1.w) << 16);
            p1.x = (unsigned)f2bf(f2.x) | ((unsigned)f2bf(f2.y) << 16);
            p1.y = (unsigned)f2bf(f2.z) | ((unsigned)f2bf(f2.w) << 16);
            p1.z = (unsigned)f2bf(f3.x) | ((unsigned)f2bf(f3.y) << 16);
            p1.w = (unsigned)f2bf(f3.z) | ((unsigned)f2bf(f3.w) << 16);
            uint4* d = reinterpret_cast<uint4*>(&Ws[nxt][wrow][wk0]);
            d[0] = p0; d[1] = p1;
        }
        wait_async0();
        __syncthreads();
    }

    // epilogue: element (v, lane, nt) -> m = wave*16 + v + 8*hl, n = nt*16 + ln
    for (int nt = 0; nt < 4; nt++) {
        int n = nBase + nt * 16 + ln;
        for (int v = 0; v < 8; v++) {
            int m = mBase + wave * 16 + v + 8 * hl;
            float val = acc[nt][v];
            if (bias) val += bias[n];
            if (flags & FLAG_GELU)
                val = 0.5f * val * (1.0f + erff(val * 0.70710678118f));
            if (flags & FLAG_SCATTER) {
                int tkn = scat_tok[m];
                if (tkn >= 0)
                    atomicAdd(outF + (size_t)tkn * N + n, scat_w[m] * val);
            } else {
                if (flags & FLAG_RES) val += residual[(size_t)m * N + n];
                if (flags & FLAG_OUTBF) outB[(size_t)m * N + n] = f2bf(val);
                else                    outF[(size_t)m * N + n] = val;
            }
        }
    }
}

// ---------------- flash attention (causal, online softmax) ----------------
// grid = (S/64, B*NH), 128 threads = 4 waves; wave owns 16 Q rows.
__global__ __launch_bounds__(128)
void flash_attn_kernel(const unsigned short* __restrict__ q,
                       const unsigned short* __restrict__ k,
                       const unsigned short* __restrict__ v,
                       unsigned short* __restrict__ ao) {
    int qTile = blockIdx.x;
    int bh = blockIdx.y;
    int b = bh >> 4, h = bh & 15;
    int tid = threadIdx.x, wave = tid >> 5, lane = tid & 31;
    int hl = lane >> 4, ln = lane & 15;

    __shared__ alignas(32) unsigned short Qs[64][64];
    __shared__ alignas(32) unsigned short Ks[64][64];
    __shared__ alignas(32) unsigned short Vt[64][64];   // Vt[d][key]
    __shared__ alignas(32) unsigned short Ps[4][16][64];

    size_t rowOff = (size_t)b * SEQ * DMODEL + (size_t)h * HEADD;
    int qBase = qTile * 64;

    {   // load Q tile: thread -> row tid>>1, 32 cols (async -> LDS)
        int r = tid >> 1, c0 = (tid & 1) * 32;
        const unsigned short* src = q + rowOff + (size_t)(qBase + r) * DMODEL + c0;
        for (int i = 0; i < 4; i++)
            async_copy_b128(lds_addr(&Qs[r][c0 + i * 8]), src + i * 8);
    }

    float rmax[8], rsum[8];
    v8f accO[4];
    const v8f z8 = {0.f,0.f,0.f,0.f,0.f,0.f,0.f,0.f};
    for (int i = 0; i < 8; i++) { rmax[i] = -3e38f; rsum[i] = 0.f; }
    for (int i = 0; i < 4; i++) accO[i] = z8;

    for (int j = 0; j <= qTile; j++) {
        int kBase = j * 64;
        wait_async0();
        __syncthreads();                       // Qs ready / Ks,Vt reusable
        {   // K tile (row-major keys x hd) via async copy
            int r = tid >> 1, c0 = (tid & 1) * 32;
            const unsigned short* src = k + rowOff + (size_t)(kBase + r) * DMODEL + c0;
            for (int i = 0; i < 4; i++)
                async_copy_b128(lds_addr(&Ks[r][c0 + i * 8]), src + i * 8);
        }
        {   // V tile, transposed into Vt[d][key]
            int d = tid >> 1, kk0 = (tid & 1) * 32;
            for (int kk = 0; kk < 32; kk++)
                Vt[d][kk0 + kk] = v[rowOff + (size_t)(kBase + kk0 + kk) * DMODEL + d];
        }
        wait_async0();
        __syncthreads();

        // scores S = Q @ K^T : preload 2 Q-frags + 8 K-frags, then 8 WMMAs
        v8f s[4];
        {
            const unsigned short* qRow = &Qs[wave * 16 + ln][0];
            v16bf qa[2];
            qa[0] = load_frag(qRow, hl);
            qa[1] = load_frag(qRow + 32, hl);
            v16bf kf[2][4];
#pragma unroll
            for (int kk = 0; kk < 2; kk++)
#pragma unroll
                for (int nt = 0; nt < 4; nt++)
                    kf[kk][nt] = load_frag(&Ks[nt * 16 + ln][kk * 32], hl);
#pragma unroll
            for (int nt = 0; nt < 4; nt++) s[nt] = z8;
#pragma unroll
            for (int kk = 0; kk < 2; kk++)
#pragma unroll
                for (int nt = 0; nt < 4; nt++)
                    s[nt] = wmma_bf16(qa[kk], kf[kk][nt], s[nt]);
        }
        // scale + causal mask + row max
        float tmax[8];
        for (int vv = 0; vv < 8; vv++) tmax[vv] = -3e38f;
        for (int nt = 0; nt < 4; nt++)
            for (int vv = 0; vv < 8; vv++) {
                float val = s[nt][vv] * 0.125f;     // 1/sqrt(64)
                int qrow = qBase + wave * 16 + vv + 8 * hl;
                int kcol = kBase + nt * 16 + ln;
                if (kcol > qrow) val = -1e30f;
                s[nt][vv] = val;
                tmax[vv] = fmaxf(tmax[vv], val);
            }
        for (int off = 1; off < 16; off <<= 1)
            for (int vv = 0; vv < 8; vv++)
                tmax[vv] = fmaxf(tmax[vv], __shfl_xor(tmax[vv], off));
        // online softmax update
        float fac[8], psum[8];
        for (int vv = 0; vv < 8; vv++) {
            float nm = fmaxf(rmax[vv], tmax[vv]);
            fac[vv] = __expf(rmax[vv] - nm);
            rmax[vv] = nm;
            psum[vv] = 0.f;
        }
        for (int nt = 0; nt < 4; nt++)
            for (int vv = 0; vv < 8; vv++) {
                float p = __expf(s[nt][vv] - rmax[vv]);
                s[nt][vv] = p;
                psum[vv] += p;
            }
        for (int off = 1; off < 16; off <<= 1)
            for (int vv = 0; vv < 8; vv++)
                psum[vv] += __shfl_xor(psum[vv], off);
        for (int vv = 0; vv < 8; vv++) rsum[vv] = rsum[vv] * fac[vv] + psum[vv];
        for (int nt = 0; nt < 4; nt++)
            for (int vv = 0; vv < 8; vv++) accO[nt][vv] *= fac[vv];
        // P -> LDS (wave-private), bf16
        for (int nt = 0; nt < 4; nt++)
            for (int vv = 0; vv < 8; vv++)
                Ps[wave][vv + 8 * hl][nt * 16 + ln] = f2bf(s[nt][vv]);
        // O += P @ V : preload 2 P-frags + 8 V-frags, then 8 WMMAs
        {
            const unsigned short* pRow = &Ps[wave][ln][0];
            v16bf pa[2];
            pa[0] = load_frag(pRow, hl);
            pa[1] = load_frag(pRow + 32, hl);
            v16bf vf[2][4];
#pragma unroll
            for (int kk = 0; kk < 2; kk++)
#pragma unroll
                for (int nt = 0; nt < 4; nt++)
                    vf[kk][nt] = load_frag(&Vt[nt * 16 + ln][kk * 32], hl);
#pragma unroll
            for (int kk = 0; kk < 2; kk++)
#pragma unroll
                for (int nt = 0; nt < 4; nt++)
                    accO[nt] = wmma_bf16(pa[kk], vf[kk][nt], accO[nt]);
        }
    }
    // finalize
    for (int nt = 0; nt < 4; nt++)
        for (int vv = 0; vv < 8; vv++) {
            int r = qBase + wave * 16 + vv + 8 * hl;
            float o = accO[nt][vv] / rsum[vv];
            ao[rowOff + (size_t)r * DMODEL + nt * 16 + ln] = f2bf(o);
        }
}

// ---------------- MoE routing ----------------
__global__ __launch_bounds__(256)
void routing_kernel(const unsigned short* __restrict__ tn,
                    const float* __restrict__ gate_w,
                    int* __restrict__ topk_idx, float* __restrict__ topk_w,
                    int* __restrict__ counts) {
    int t = blockIdx.x * 8 + (threadIdx.x >> 5);
    int lane = threadIdx.x & 31;
    if (t >= NTOK) return;
    float acc[NEXP];
    for (int e = 0; e < NEXP; e++) acc[e] = 0.f;
    const unsigned short* row = tn + (size_t)t * DMODEL;
    for (int i = lane; i < DMODEL; i += 32) {
        float xv = bf2f(row[i]);
        for (int e = 0; e < NEXP; e++) acc[e] += xv * gate_w[e * DMODEL + i];
    }
    for (int off = 16; off; off >>= 1)
        for (int e = 0; e < NEXP; e++) acc[e] += __shfl_xor(acc[e], off);
    if (lane == 0) {
        int i1 = 0; float m1 = acc[0];
        for (int e = 1; e < NEXP; e++) if (acc[e] > m1) { m1 = acc[e]; i1 = e; }
        int i2 = (i1 == 0) ? 1 : 0; float m2 = acc[i2];
        for (int e = 0; e < NEXP; e++)
            if (e != i1 && acc[e] > m2) { m2 = acc[e]; i2 = e; }
        float e2 = __expf(m2 - m1);
        float w1 = 1.0f / (1.0f + e2);
        float w2 = e2 * w1;
        topk_idx[t * 2]     = i1;  topk_idx[t * 2 + 1] = i2;
        topk_w[t * 2]       = w1;  topk_w[t * 2 + 1]   = w2;
        atomicAdd(&counts[i1], 1);
        atomicAdd(&counts[i2], 1);
    }
}

__global__ void zero_ints_kernel(int* counts, int* cursor) {
    int i = threadIdx.x;
    if (i < NEXP) { counts[i] = 0; cursor[i] = 0; }
}

__global__ void offsets_kernel(const int* __restrict__ counts, int* __restrict__ off) {
    if (threadIdx.x == 0) {
        int a = 0; off[0] = 0;
        for (int e = 0; e < NEXP; e++) {
            a += ((counts[e] + 127) >> 7) << 7;      // 128-align segments
            off[e + 1] = a;
        }
    }
}

__global__ __launch_bounds__(256)
void fill_pairs_kernel(const int* __restrict__ off, int* __restrict__ pair_token,
                       int* __restrict__ pair_expert, float* __restrict__ pair_w) {
    int p = blockIdx.x * blockDim.x + threadIdx.x;
    if (p >= PAIR_CAP) return;
    pair_token[p] = -1;
    pair_w[p] = 0.f;
    int pe = NEXP - 1;
    for (int e = 0; e < NEXP; e++)
        if (p < off[e + 1]) { pe = e; break; }
    pair_expert[p] = pe;
}

__global__ __launch_bounds__(256)
void scatter_pairs_kernel(const int* __restrict__ topk_idx, const float* __restrict__ topk_w,
                          const int* __restrict__ off, int* __restrict__ cursor,
                          int* __restrict__ pair_token, float* __restrict__ pair_w) {
    int t = blockIdx.x * blockDim.x + threadIdx.x;
    if (t >= NTOK) return;
    for (int kk = 0; kk < TOPK; kk++) {
        int e = topk_idx[t * 2 + kk];
        int pos = off[e] + atomicAdd(&cursor[e], 1);
        pair_token[pos] = t;
        pair_w[pos] = topk_w[t * 2 + kk];
    }
}

// ---------------- host-side launch ----------------
extern "C" void kernel_launch(void* const* d_in, const int* in_sizes, int n_in,
                              void* d_out, int out_size, void* d_ws, size_t ws_size,
                              hipStream_t stream) {
    (void)in_sizes; (void)n_in; (void)out_size; (void)ws_size;
    const float* x      = (const float*)d_in[0];
    const float* anw    = (const float*)d_in[1];
    const float* wq     = (const float*)d_in[2];
    const float* wk     = (const float*)d_in[3];
    const float* wv     = (const float*)d_in[4];
    const float* wo     = (const float*)d_in[5];
    const float* mnw    = (const float*)d_in[6];
    const float* gate_w = (const float*)d_in[7];
    const float* w1     = (const float*)d_in[8];
    const float* b1     = (const float*)d_in[9];
    const float* w2     = (const float*)d_in[10];
    const float* b2     = (const float*)d_in[11];
    float* out = (float*)d_out;

    // workspace carve-up (~56.2 MB); buffers aliased across disjoint lifetimes
    char* ws = (char*)d_ws;
    size_t o = 0;
    const size_t actB = (size_t)NTOK * DMODEL * sizeof(unsigned short);   // 4 MB
    unsigned short* buf0 = (unsigned short*)(ws + o); o += actB;  // tnA, then ao
    unsigned short* buf1 = (unsigned short*)(ws + o); o += actB;  // qb, then tnM
    unsigned short* kb   = (unsigned short*)(ws + o); o += actB;
    unsigned short* vb   = (unsigned short*)(ws + o); o += actB;
    unsigned short* hid  = (unsigned short*)(ws + o);
    o += (size_t)PAIR_CAP * HHID * sizeof(unsigned short);                // 40 MB
    int*   pair_token  = (int*)(ws + o);   o += PAIR_CAP * sizeof(int);
    int*   pair_expert = (int*)(ws + o);   o += PAIR_CAP * sizeof(int);
    float* pair_w      = (float*)(ws + o); o += PAIR_CAP * sizeof(float);
    int*   topk_idx    = (int*)(ws + o);   o += (size_t)NTOK * 2 * sizeof(int);
    float* topk_w      = (float*)(ws + o); o += (size_t)NTOK * 2 * sizeof(float);
    int*   counts      = (int*)(ws + o);   o += 64;
    int*   eoff        = (int*)(ws + o);   o += 64;
    int*   cursor      = (int*)(ws + o);   o += 64;

    unsigned short* tnA = buf0;
    unsigned short* qb  = buf1;
    unsigned short* ao  = buf0;   // tnA dead after QKV projections
    unsigned short* tnM = buf1;   // qb dead after flash attention

    // 1) attention RMSNorm -> bf16
    rmsnorm_kernel<<<NTOK, 256, 0, stream>>>(x, anw, tnA, DMODEL);

    // 2) QKV projections (bf16 WMMA, f32 weights streamed + converted in LDS)
    dim3 gProj(DMODEL / 64, NTOK / 128);
    gemm_bf16_wmma<<<gProj, 256, 0, stream>>>(tnA, DMODEL, wq, DMODEL, nullptr, nullptr,
        nullptr, qb, NTOK, DMODEL, DMODEL, nullptr, nullptr, 0, 0, nullptr, nullptr, FLAG_OUTBF);
    gemm_bf16_wmma<<<gProj, 256, 0, stream>>>(tnA, DMODEL, wk, DMODEL, nullptr, nullptr,
        nullptr, kb, NTOK, DMODEL, DMODEL, nullptr, nullptr, 0, 0, nullptr, nullptr, FLAG_OUTBF);
    gemm_bf16_wmma<<<gProj, 256, 0, stream>>>(tnA, DMODEL, wv, DMODEL, nullptr, nullptr,
        nullptr, vb, NTOK, DMODEL, DMODEL, nullptr, nullptr, 0, 0, nullptr, nullptr, FLAG_OUTBF);

    // 3) RoPE on q,k
    rope_kernel<<<(NTOK * NHEADS * 32) / 256, 256, 0, stream>>>(qb, kb);

    // 4) causal flash attention  (ao aliases tnA)
    flash_attn_kernel<<<dim3(SEQ / 64, BATCH * NHEADS), 128, 0, stream>>>(qb, kb, vb, ao);

    // 5) output projection + residual -> h (written to d_out, f32)
    gemm_bf16_wmma<<<gProj, 256, 0, stream>>>(ao, DMODEL, wo, DMODEL, nullptr, x,
        out, nullptr, NTOK, DMODEL, DMODEL, nullptr, nullptr, 0, 0, nullptr, nullptr, FLAG_RES);

    // 6) MoE RMSNorm on h  (tnM aliases qb)
    rmsnorm_kernel<<<NTOK, 256, 0, stream>>>(out, mnw, tnM, DMODEL);

    // 7) routing: logits -> top-2 + softmax, per-expert compaction
    zero_ints_kernel<<<1, 64, 0, stream>>>(counts, cursor);
    routing_kernel<<<NTOK / 8, 256, 0, stream>>>(tnM, gate_w, topk_idx, topk_w, counts);
    offsets_kernel<<<1, 1, 0, stream>>>(counts, eoff);
    fill_pairs_kernel<<<(PAIR_CAP + 255) / 256, 256, 0, stream>>>(eoff, pair_token, pair_expert, pair_w);
    scatter_pairs_kernel<<<(NTOK + 255) / 256, 256, 0, stream>>>(topk_idx, topk_w, eoff, cursor,
                                                                 pair_token, pair_w);

    // 8) MoE pass 1: hid = gelu(tn[tok] @ w1[e]^T + b1[e])  (grouped by expert)
    gemm_bf16_wmma<<<dim3(HHID / 64, PAIR_CAP / 128), 256, 0, stream>>>(
        tnM, DMODEL, w1, DMODEL, b1, nullptr, nullptr, hid,
        PAIR_CAP, HHID, DMODEL,
        pair_token /*gather A*/, pair_expert, (size_t)HHID * DMODEL, (size_t)HHID,
        nullptr, nullptr, FLAG_GELU | FLAG_OUTBF);

    // 9) MoE pass 2: d_out[tok] += rw * (hid @ w2[e]^T + b2[e])
    gemm_bf16_wmma<<<dim3(DMODEL / 64, PAIR_CAP / 128), 256, 0, stream>>>(
        hid, HHID, w2, HHID, b2, nullptr, out, nullptr,
        PAIR_CAP, DMODEL, HHID,
        nullptr, pair_expert, (size_t)DMODEL * HHID, (size_t)DMODEL,
        pair_token, pair_w, FLAG_SCATTER);
}
